// TransformerEncoderLayerWithAttnOut_38465727103854
// MI455X (gfx1250) — compile-verified, matrix operand order checked by reading, not run on hardware
//
#include <hip/hip_runtime.h>
#include <hip/hip_bf16.h>

typedef __attribute__((ext_vector_type(16))) _Float16 v16h;
typedef __attribute__((ext_vector_type(8)))  _Float16 v8h;
typedef __attribute__((ext_vector_type(4)))  _Float16 v4h;
typedef __attribute__((ext_vector_type(8)))  float    v8f;
typedef __attribute__((ext_vector_type(4)))  unsigned int u32x4;
typedef __attribute__((ext_vector_type(8)))  int      i32x8;
typedef __attribute__((ext_vector_type(4)))  int      i32x4;

#define Bn 4
#define Tt 1024
#define Dd 1024
#define Hh 16
#define HD 64
#define NROWS (Bn * Tt)   // 4096

#if __has_builtin(__builtin_amdgcn_tensor_load_to_lds) && __has_builtin(__builtin_amdgcn_s_wait_tensorcnt)
#define HAVE_TDM 1
#else
#define HAVE_TDM 0
#endif

// ---------------------------------------------------------------------------
// WMMA 16x16x32 f16 fragment load from a row-major f16 buffer.
// Per CDNA5 ISA 16-bit A layout (7.12.2): lanes 0-15 hold row (lane&15),
// K elements {0..7, 16..23}; lanes 16-31 hold K {8..15, 24..31}.
// Both halves are contiguous 16B chunks -> two b128 loads per lane.
// ---------------------------------------------------------------------------
__device__ __forceinline__ v16h load_frag(const _Float16* base, int pitch, int lane) {
    int r  = lane & 15;
    int ko = (lane < 16) ? 0 : 8;
    const _Float16* p = base + r * pitch + ko;
    v8h lo = *(const v8h*)(p);
    v8h hi = *(const v8h*)(p + 16);
    v16h out;
#pragma unroll
    for (int i = 0; i < 8; ++i) { out[i] = lo[i]; out[i + 8] = hi[i]; }
    return out;
}

__device__ __forceinline__ v8f wmma_f16(v16h a, v16h b, v8f c) {
    return __builtin_amdgcn_wmma_f32_16x16x32_f16(false, a, false, b, (short)0, c, false, false);
}

// ---------------------------------------------------------------------------
// f32 -> f16 conversion, 4 elements/thread (activations)
// ---------------------------------------------------------------------------
__global__ void __launch_bounds__(256) cvt_f32_f16_kernel(const float* __restrict__ in,
                                                          _Float16* __restrict__ out, int n) {
    int i = (blockIdx.x * 256 + threadIdx.x) * 4;
    if (i < n) {
        float4 v = *(const float4*)(in + i);
        v4h o;
        o[0] = (_Float16)v.x; o[1] = (_Float16)v.y;
        o[2] = (_Float16)v.z; o[3] = (_Float16)v.w;
        *(v4h*)(out + i) = o;
    }
}

// ---------------------------------------------------------------------------
// Transposing f32 -> f16 conversion for weights: in[R,C] f32 -> out[C,R] f16.
// 32x32 LDS tile; R,C multiples of 32. Makes every GEMM B-operand K-contiguous
// so the Tensor Data Mover can stage it with a plain 2D descriptor.
// ---------------------------------------------------------------------------
__global__ void __launch_bounds__(256) cvt_transpose_kernel(const float* __restrict__ in,
                                                            _Float16* __restrict__ out,
                                                            int R, int C) {
    __shared__ _Float16 tile[32][33];
    int tx = threadIdx.x & 31, ty = threadIdx.x >> 5;   // ty: 0..7
    int r0 = blockIdx.y * 32, c0 = blockIdx.x * 32;
#pragma unroll
    for (int i = 0; i < 4; ++i) {
        int r = ty * 4 + i;
        tile[r][tx] = (_Float16)in[(long long)(r0 + r) * C + c0 + tx];
    }
    __syncthreads();
#pragma unroll
    for (int i = 0; i < 4; ++i) {
        int c = ty * 4 + i;
        out[(long long)(c0 + c) * R + r0 + tx] = tile[tx][c];
    }
}

// ---------------------------------------------------------------------------
// Tiled WMMA GEMM: C[M,N] = A16[M,K] @ BT16[N,K]^T + bias, optional exact GELU.
// Block tile 128x128, 256 threads = 8 waves, each wave 32x64 (2x4 WMMA tiles),
// BK=64 -> 16 WMMAs / wave / K-tile. Both operand tiles are 128 rows x 64
// K-contiguous f16 and are staged by the TDM (one 2D descriptor each, LDS pad
// 4 DWORDs per 32 -> pitch 72), double-buffered so the DMA for tile i+1 runs
// under the WMMAs of tile i, one barrier per K-tile. Fallback: b128 copies.
// ---------------------------------------------------------------------------
#define BM 128
#define BN 128
#define BK 64
#define LDP (BK + 8)            // f16 pitch 72 -> 144B rows (16B aligned)
#define TILE_ELEMS (128 * LDP)  // one staged operand tile in LDS
#define GEMM_SMEM (4 * TILE_ELEMS * 2)  // 2 operands x 2 buffers, bytes

#if HAVE_TDM
// 2D TDM load: tile = 128 rows x BK f16, row stride = rowStride elems,
// dest LDS padded 4 DWORDs per 32 DWORDs (pitch 72 f16).
__device__ __forceinline__ void tdm_load_tile(_Float16* lds, const _Float16* gsrc,
                                              int rowStride) {
    unsigned long long ga = (unsigned long long)(size_t)gsrc;
    unsigned lds_off = (unsigned)(size_t)lds;   // generic addr low bits = LDS offset
    unsigned td0 = (unsigned)rowStride;         // tensor_dim0 (OOB bound)
    unsigned td1 = 1u << 20;                    // tensor_dim1: generous bound
    unsigned long long s0 = (unsigned long long)rowStride;

    u32x4 g0 = {};
    g0[0] = 1u;                                             // count=1, user mode
    g0[1] = lds_off;                                        // lds_addr (bytes)
    g0[2] = (unsigned)ga;                                   // global_addr[31:0]
    g0[3] = (unsigned)((ga >> 32) & 0x01FFFFFFu) | (2u << 30);  // [56:32] | type=2

    i32x8 g1 = {};
    g1[0] = (int)((1u << 16)        // data_size = 1 -> 2 bytes
                | (1u << 20)        // pad_enable
                | (4u << 22)        // pad_interval: 32 DWORDs
                | (3u << 25));      // pad_amount: 4 DWORDs (8 f16)
    g1[1] = (int)((td0 & 0xFFFFu) << 16);                       // tensor_dim0 lo
    g1[2] = (int)((td0 >> 16) | ((td1 & 0xFFFFu) << 16));       // dim0 hi | dim1 lo
    g1[3] = (int)((td1 >> 16) | ((unsigned)BK << 16));          // dim1 hi | tile_dim0
    g1[4] = (int)((unsigned)BM);                                // tile_dim1 | tile_dim2=0
    g1[5] = (int)(unsigned)(s0 & 0xFFFFFFFFu);                  // dim0_stride lo
    g1[6] = (int)(unsigned)((s0 >> 32) & 0xFFFFu);              // dim0_stride hi
    g1[7] = 0;

    i32x4 z4 = {};
#if __has_include(<hip/amd_detail/amd_gfx1250_TDM.h>)
    i32x8 z8 = {};
    __builtin_amdgcn_tensor_load_to_lds(g0, g1, z4, z4, z8, 0);
#else
    __builtin_amdgcn_tensor_load_to_lds(g0, g1, z4, z4, 0);
#endif
}
#endif

// Stage one 128x64 f16 tile (A or B^T) into LDS at `dst` (pitch LDP).
__device__ __forceinline__ void stage_tile(_Float16* dst, const _Float16* gsrc,
                                           int ld, int tid, int wave) {
#if HAVE_TDM
    (void)tid;
    if (wave == 0) tdm_load_tile(dst, gsrc, ld);
#else
    (void)wave;
#pragma unroll
    for (int t = 0; t < 4; ++t) {
        int idx = (tid + t * 256) * 8;
        int r = idx >> 6, c = idx & 63;
        v8h v = *(const v8h*)(gsrc + (long long)r * ld + c);
        *(v8h*)(dst + r * LDP + c) = v;
    }
#endif
}

__global__ void __launch_bounds__(256) gemm_f16_wmma_kernel(
    const _Float16* __restrict__ A,   // [M,K]
    const _Float16* __restrict__ BT,  // [N,K] (pre-transposed weights)
    const float* __restrict__ bias, float* __restrict__ C32,
    _Float16* __restrict__ C16, int M, int N, int K, int flags) {

    extern __shared__ char smem[];
    _Float16* sA  = (_Float16*)smem;                  // [2][128][LDP]
    _Float16* sBT = (_Float16*)smem + 2 * TILE_ELEMS; // [2][128][LDP]

    int tid  = threadIdx.x;
    int lane = tid & 31;
    int wave = tid >> 5;
    int bm = blockIdx.y * BM;
    int bn = blockIdx.x * BN;
    int wm = (wave >> 1) * 32;          // 0 / 32 / 64 / 96
    int wn = (wave & 1) * 64;           // 0 / 64

    const _Float16* aTile = A  + (long long)bm * K;
    const _Float16* bTile = BT + (long long)bn * K;

    v8f acc[2][4] = {};

    int nt = K / BK;
    // prologue: stage tile 0 into buffer 0
    stage_tile(sA, aTile, K, tid, wave);
    stage_tile(sBT, bTile, K, tid, wave);
#if HAVE_TDM
    if (wave == 0) __builtin_amdgcn_s_wait_tensorcnt(0);
#endif
    __syncthreads();

    for (int i = 0; i < nt; ++i) {
        // stage tile i+1 into the other buffer (TDM overlaps with WMMAs below)
        if (i + 1 < nt) {
            int buf = (i + 1) & 1;
            int k1 = (i + 1) * BK;
            stage_tile(sA  + buf * TILE_ELEMS, aTile + k1, K, tid, wave);
            stage_tile(sBT + buf * TILE_ELEMS, bTile + k1, K, tid, wave);
        }
        // compute tile i
        _Float16* cA  = sA  + (i & 1) * TILE_ELEMS;
        _Float16* cBT = sBT + (i & 1) * TILE_ELEMS;
#pragma unroll
        for (int ks = 0; ks < BK; ks += 32) {
            v16h af[2], bf[4];
#pragma unroll
            for (int ii = 0; ii < 2; ++ii)
                af[ii] = load_frag(cA + (wm + ii * 16) * LDP + ks, LDP, lane);
#pragma unroll
            for (int j = 0; j < 4; ++j)
                bf[j] = load_frag(cBT + (wn + j * 16) * LDP + ks, LDP, lane);
#pragma unroll
            for (int ii = 0; ii < 2; ++ii)
#pragma unroll
                for (int j = 0; j < 4; ++j)
                    acc[ii][j] = wmma_f16(af[ii], bf[j], acc[ii][j]);
        }
#if HAVE_TDM
        if (wave == 0) __builtin_amdgcn_s_wait_tensorcnt(0);
#endif
        __syncthreads();
    }

    // --- epilogue: C layout (7.12.2): lane<16 -> M=vgpr, lane>=16 -> M=vgpr+8
    int n  = lane & 15;
    int r0 = (lane < 16) ? 0 : 8;
#pragma unroll
    for (int i = 0; i < 2; ++i)
#pragma unroll
        for (int j = 0; j < 4; ++j) {
            int gn = bn + wn + j * 16 + n;
            float bv = bias ? bias[gn] : 0.0f;
#pragma unroll
            for (int r = 0; r < 8; ++r) {
                int gm = bm + wm + i * 16 + r0 + r;
                float x = acc[i][j][r] + bv;
                if (flags & 1)  // exact GELU
                    x = 0.5f * x * (1.0f + erff(x * 0.70710678118654752f));
                long long o = (long long)gm * N + gn;
                if (C32) C32[o] = x;
                if (C16) C16[o] = (_Float16)x;
            }
        }
}

// ---------------------------------------------------------------------------
// Attention: one workgroup per (b*H+h, 32 query rows).
// Phase 1: logits = QK^T/temp + rel_bias via WMMA (Q,K frags straight from global)
// Phase 2: wave-parallel softmax; writes attn_weights (f32, d_out) + probs (f16, LDS)
// Phase 3: P @ V via WMMA with per-32-chunk V^T staged in LDS.
// Dynamic LDS: logits f32 [32][1028] | probs f16 [32][1032] | vT f16 [64][40]
// ---------------------------------------------------------------------------
#define LP (Tt + 4)
#define PP (Tt + 8)
#define VP 40
#define ATTN_SMEM (32 * LP * 4 + 32 * PP * 2 + HD * VP * 2)

__global__ void __launch_bounds__(256) attn_kernel(
    const _Float16* __restrict__ qkv,       // [B*T, 3*D] f16 (q|k|v per head packed)
    const float* __restrict__ rel_bias,     // [H, 2T-1]
    const float* __restrict__ temp_ptr,     // [1]
    float* __restrict__ attnW,              // [B,H,T,T]
    _Float16* __restrict__ attnout) {       // [B*T, D] f16

    extern __shared__ char smem[];
    float*    slog  = (float*)smem;
    _Float16* sprob = (_Float16*)(slog + 32 * LP);
    _Float16* svt   = sprob + 32 * PP;

    int tid = threadIdx.x, lane = tid & 31, wave = tid >> 5;
    int qb = blockIdx.x * 32;
    int bh = blockIdx.y;
    int b  = bh >> 4, h = bh & 15;
    float invt = 1.0f / fmaxf(temp_ptr[0], 0.01f);

    const int RS = 3 * Dd;  // qkv row stride in f16 elements
    const _Float16* qp = qkv + (long long)(b * Tt) * RS + h * HD;
    const _Float16* kp = qp + Dd;
    const _Float16* vp = qp + 2 * Dd;

    // preload Q fragments [mtile][khalf] (HD=64 -> two K=32 steps)
    v16h qf[2][2];
#pragma unroll
    for (int mt = 0; mt < 2; ++mt)
#pragma unroll
        for (int kh = 0; kh < 2; ++kh)
            qf[mt][kh] = load_frag(qp + (long long)(qb + mt * 16) * RS + kh * 32, RS, lane);

    // ---- Phase 1: logits ----
    int n  = lane & 15;
    int r0 = (lane < 16) ? 0 : 8;
    for (int it = 0; it < 8; ++it) {
        int jt = wave + it * 8;                 // 64 column tiles of 16
        v16h kf0 = load_frag(kp + (long long)(jt * 16) * RS +  0, RS, lane);
        v16h kf1 = load_frag(kp + (long long)(jt * 16) * RS + 32, RS, lane);
#pragma unroll
        for (int mt = 0; mt < 2; ++mt) {
            v8f acc = {};
            acc = wmma_f16(qf[mt][0], kf0, acc);
            acc = wmma_f16(qf[mt][1], kf1, acc);
            int kcol = jt * 16 + n;
#pragma unroll
            for (int r = 0; r < 8; ++r) {
                int m = mt * 16 + r0 + r;
                int qrow = qb + m;
                float l = acc[r] * invt +
                          rel_bias[h * (2 * Tt - 1) + (kcol - qrow + Tt - 1)];
                slog[m * LP + kcol] = l;
            }
        }
    }
    __syncthreads();

    // ---- Phase 2: softmax (wave handles 4 rows) ----
    for (int rr = 0; rr < 4; ++rr) {
        int m = wave * 4 + rr;
        float* row = slog + m * LP;
        float mx = -3.4e38f;
        for (int i = lane; i < Tt; i += 32) mx = fmaxf(mx, row[i]);
#pragma unroll
        for (int o = 16; o > 0; o >>= 1) mx = fmaxf(mx, __shfl_xor(mx, o, 32));
        float sum = 0.0f;
        for (int i = lane; i < Tt; i += 32) {
            float e = __expf(row[i] - mx);
            row[i] = e; sum += e;
        }
#pragma unroll
        for (int o = 16; o > 0; o >>= 1) sum += __shfl_xor(sum, o, 32);
        float inv = 1.0f / sum;
        long long wbase = ((long long)bh * Tt + (qb + m)) * Tt;
        for (int i = lane; i < Tt; i += 32) {
            float p = row[i] * inv;
            attnW[wbase + i] = p;
            sprob[m * PP + i] = (_Float16)p;
        }
    }
    __syncthreads();

    // ---- Phase 3: P @ V (each wave one 16x16 out tile; mt=wave>>2, nt=wave&3) ----
    int mt = wave >> 2, nt = wave & 3;
    v8f acc = {};
    for (int kb = 0; kb < Tt / 32; ++kb) {
        // stage V^T chunk: [64 hd][32 t]
        {
            int idx = tid * 8;
            int t = idx >> 6, hd = idx & 63;
            v8h v = *(const v8h*)(vp + (long long)(kb * 32 + t) * RS + hd);
#pragma unroll
            for (int i = 0; i < 8; ++i) svt[(hd + i) * VP + t] = v[i];
            if (kb + 1 < Tt / 32)
                __builtin_prefetch(vp + (long long)((kb + 1) * 32 + t) * RS + hd, 0, 3);
        }
        __syncthreads();
        v16h af = load_frag(sprob + mt * 16 * PP + kb * 32, PP, lane);
        v16h bf = load_frag(svt + nt * 16 * VP, VP, lane);
        acc = wmma_f16(af, bf, acc);
        __syncthreads();
    }
#pragma unroll
    for (int r = 0; r < 8; ++r) {
        int m = mt * 16 + r0 + r;
        long long o = (long long)(b * Tt + qb + m) * Dd + h * HD + nt * 16 + n;
        attnout[o] = (_Float16)acc[r];
    }
}

// ---------------------------------------------------------------------------
// Residual + LayerNorm: one block per row (D=1024, 256 threads x 4 cols)
// val = res[c] + scale * a[c];  out = LN(val)*gamma + beta
// ---------------------------------------------------------------------------
__global__ void __launch_bounds__(256) addln_kernel(
    const float* __restrict__ a, const float* __restrict__ res,
    const float* __restrict__ scale_ptr,
    const float* __restrict__ gamma, const float* __restrict__ beta,
    float* __restrict__ out32, _Float16* __restrict__ out16) {

    __shared__ float red0[256];
    __shared__ float red1[256];
    int row = blockIdx.x, tid = threadIdx.x;
    float s = scale_ptr ? scale_ptr[0] : 1.0f;
    const float* pa = a   + (long long)row * Dd;
    const float* pr = res + (long long)row * Dd;

    float v[4]; float sum = 0.0f, sq = 0.0f;
#pragma unroll
    for (int i = 0; i < 4; ++i) {
        int c = tid + i * 256;
        float x = pr[c] + s * pa[c];
        v[i] = x; sum += x; sq += x * x;
    }
    red0[tid] = sum; red1[tid] = sq;
    __syncthreads();
    for (int o = 128; o > 0; o >>= 1) {
        if (tid < o) { red0[tid] += red0[tid + o]; red1[tid] += red1[tid + o]; }
        __syncthreads();
    }
    float mu  = red0[0] * (1.0f / Dd);
    float var = red1[0] * (1.0f / Dd) - mu * mu;
    float rinv = rsqrtf(var + 1e-5f);
#pragma unroll
    for (int i = 0; i < 4; ++i) {
        int c = tid + i * 256;
        float y = (v[i] - mu) * rinv * gamma[c] + beta[c];
        long long o = (long long)row * Dd + c;
        if (out32) out32[o] = y;
        if (out16) out16[o] = (_Float16)y;
    }
}

// ---------------------------------------------------------------------------
extern "C" void kernel_launch(void* const* d_in, const int* in_sizes, int n_in,
                              void* d_out, int out_size, void* d_ws, size_t ws_size,
                              hipStream_t stream) {
    (void)in_sizes; (void)n_in; (void)out_size; (void)ws_size;

    const float* src      = (const float*)d_in[0];
    const float* Wqkv     = (const float*)d_in[1];
    const float* bqkv     = (const float*)d_in[2];
    const float* Wo       = (const float*)d_in[3];
    const float* bo       = (const float*)d_in[4];
    const float* rel_bias = (const float*)d_in[5];
    const float* temp     = (const float*)d_in[6];
    const float* W1       = (const float*)d_in[7];
    const float* b1       = (const float*)d_in[8];
    const float* W2       = (const float*)d_in[9];
    const float* b2       = (const float*)d_in[10];
    const float* ff_scale = (const float*)d_in[11];
    const float* gamma1   = (const float*)d_in[12];
    const float* beta1    = (const float*)d_in[13];
    const float* gamma2   = (const float*)d_in[14];
    const float* beta2    = (const float*)d_in[15];

    float* out   = (float*)d_out;                       // [B,T,D]
    float* attnW = out + (long long)NROWS * Dd;         // [B,H,T,T]

    // scratch carve-out
    char* ws = (char*)d_ws;
    size_t off = 0;
    auto alloc = [&](size_t bytes) -> char* {
        char* p = ws + off;
        off += (bytes + 255) & ~(size_t)255;
        return p;
    };
    _Float16* src16     = (_Float16*)alloc((size_t)NROWS * Dd * 2);
    _Float16* WqkvT16   = (_Float16*)alloc((size_t)Dd * 3 * Dd * 2);   // [3D, D]
    _Float16* WoT16     = (_Float16*)alloc((size_t)Dd * Dd * 2);       // [D, D]
    _Float16* W1T16     = (_Float16*)alloc((size_t)Dd * 4 * Dd * 2);   // [4D, D]
    _Float16* W2T16     = (_Float16*)alloc((size_t)4 * Dd * Dd * 2);   // [D, 4D]
    _Float16* qkv16     = (_Float16*)alloc((size_t)NROWS * 3 * Dd * 2);
    _Float16* attnout16 = (_Float16*)alloc((size_t)NROWS * Dd * 2);
    float*    proj32    = (float*)   alloc((size_t)NROWS * Dd * 4);
    float*    x32       = (float*)   alloc((size_t)NROWS * Dd * 4);
    _Float16* x16       = (_Float16*)alloc((size_t)NROWS * Dd * 2);
    _Float16* ff116     = (_Float16*)alloc((size_t)NROWS * 4 * Dd * 2);
    float*    ff232     = (float*)   alloc((size_t)NROWS * Dd * 4);

    hipFuncSetAttribute((const void*)attn_kernel,
                        hipFuncAttributeMaxDynamicSharedMemorySize, ATTN_SMEM);
    hipFuncSetAttribute((const void*)gemm_f16_wmma_kernel,
                        hipFuncAttributeMaxDynamicSharedMemorySize, GEMM_SMEM);

    // 1. conversions: src straight, weights transposed to [N,K]
    cvt_f32_f16_kernel<<<NROWS * Dd / 1024, 256, 0, stream>>>(src, src16, NROWS * Dd);
    cvt_transpose_kernel<<<dim3(3 * Dd / 32, Dd / 32), 256, 0, stream>>>(Wqkv, WqkvT16, Dd, 3 * Dd);
    cvt_transpose_kernel<<<dim3(Dd / 32, Dd / 32), 256, 0, stream>>>(Wo, WoT16, Dd, Dd);
    cvt_transpose_kernel<<<dim3(4 * Dd / 32, Dd / 32), 256, 0, stream>>>(W1, W1T16, Dd, 4 * Dd);
    cvt_transpose_kernel<<<dim3(Dd / 32, 4 * Dd / 32), 256, 0, stream>>>(W2, W2T16, 4 * Dd, Dd);

    // 2. QKV projection: [4096,1024] @ [1024,3072] + bqkv -> qkv16
    gemm_f16_wmma_kernel<<<dim3(3 * Dd / BN, NROWS / BM), 256, GEMM_SMEM, stream>>>(
        src16, WqkvT16, bqkv, nullptr, qkv16, NROWS, 3 * Dd, Dd, 0);

    // 3. Attention (writes attn_weights to d_out and attnout16)
    attn_kernel<<<dim3(Tt / 32, Bn * Hh), 256, ATTN_SMEM, stream>>>(
        qkv16, rel_bias, temp, attnW, attnout16);

    // 4. Output projection: [4096,1024] @ [1024,1024] + bo -> proj32
    gemm_f16_wmma_kernel<<<dim3(Dd / BN, NROWS / BM), 256, GEMM_SMEM, stream>>>(
        attnout16, WoT16, bo, proj32, nullptr, NROWS, Dd, Dd, 0);

    // 5. x = LN(src + proj) -> x32, x16
    addln_kernel<<<NROWS, 256, 0, stream>>>(proj32, src, nullptr, gamma1, beta1, x32, x16);

    // 6. FFN up: [4096,1024] @ [1024,4096] + b1, GELU -> ff116
    gemm_f16_wmma_kernel<<<dim3(4 * Dd / BN, NROWS / BM), 256, GEMM_SMEM, stream>>>(
        x16, W1T16, b1, nullptr, ff116, NROWS, 4 * Dd, Dd, 1);

    // 7. FFN down: [4096,4096] @ [4096,1024] + b2 -> ff232
    gemm_f16_wmma_kernel<<<dim3(Dd / BN, NROWS / BM), 256, GEMM_SMEM, stream>>>(
        ff116, W2T16, b2, ff232, nullptr, NROWS, Dd, 4 * Dd, 0);

    // 8. out = LN(x + ff_scale * ff) -> d_out
    addln_kernel<<<NROWS, 256, 0, stream>>>(ff232, x32, ff_scale, gamma2, beta2, out, nullptr);
}
